// MultiHeadAttention1d_43439299231791
// MI455X (gfx1250) — matrix operand order repaired and checked by reading, not compiled, not run on hardware
//
#include <hip/hip_runtime.h>

// Problem constants (match the JAX reference).
#define BB  4
#define CC  256
#define NN  2048
#define HH  8
#define DD  256
#define OCC 2048

typedef __attribute__((ext_vector_type(16))) __bf16    v16bf;
typedef __attribute__((ext_vector_type(8)))  float     v8f;
typedef __attribute__((ext_vector_type(4)))  unsigned  u32x4;
typedef __attribute__((ext_vector_type(2)))  unsigned  u32x2;
typedef __attribute__((ext_vector_type(8)))  int       i32x8;
typedef __attribute__((ext_vector_type(4)))  int       i32x4;
typedef __attribute__((ext_vector_type(4)))  float     f32x4;

union Frag {
  v16bf v;
  u32x4 q[2];
  unsigned u[8];
  unsigned short s[16];
};

__device__ __forceinline__ unsigned short f2bf(float f) {
  union { float f; unsigned u; } cv; cv.f = f;
  unsigned r = cv.u + 0x7FFFu + ((cv.u >> 16) & 1u);  // round-to-nearest-even
  return (unsigned short)(r >> 16);
}

__device__ __forceinline__ unsigned pack2bf(float a, float b) {
  return (unsigned)f2bf(a) | ((unsigned)f2bf(b) << 16);
}

// A-matrix (16x32, 16-bit): lane's halves are k = [8*hi, 8*hi+8) and
// [16+8*hi, 16+8*hi+8)  ->  two aligned 16B loads from a K-contiguous row.
__device__ __forceinline__ void loadA(Frag& f, const unsigned short* base, int hi) {
  f.q[0] = *(const u32x4*)(base + 8 * hi);
  f.q[1] = *(const u32x4*)(base + 16 + 8 * hi);
}
// B-matrix (32x16, 16-bit): lane's halves are k = [8*hi, 8*hi+8) and
// [16+8*hi, 16+8*hi+8)  ->  two aligned 16B loads from a K-contiguous column.
__device__ __forceinline__ void loadB(Frag& f, const unsigned short* base, int hi) {
  f.q[0] = *(const u32x4*)(base + 8 * hi);
  f.q[1] = *(const u32x4*)(base + 16 + 8 * hi);
}

// ---------------------------------------------------------------------------
// Tensor Data Mover: 2-D tile load Global->LDS (CDNA5 ISA).
// D# group0: {count=1 | lds_addr | global_addr[56:0] | type=2}
// D# group1: {data_size=2B, pad ctl, tensor_dim0/1, tile_dim0/1, dim0_stride}
// Groups 2/3 zero (<=2D tensor). Tracked by TENSORcnt.
// ---------------------------------------------------------------------------
__device__ __forceinline__ void tdm_load_2d(
    unsigned lds_addr, const void* gptr,
    unsigned td0, unsigned td1, unsigned tile0, unsigned tile1,
    unsigned long long stride0, unsigned padi, unsigned pada, bool pad) {
  const unsigned long long ga = (unsigned long long)gptr;
  u32x4 g0;
  g0[0] = 1u;                                               // count=1 (user D#)
  g0[1] = lds_addr;                                         // LDS byte offset
  g0[2] = (unsigned)ga;                                     // global_addr[31:0]
  g0[3] = (unsigned)((ga >> 32) & 0x01FFFFFFull) | (2u << 30);  // [56:32]|type=2
  i32x8 g1;
  g1[0] = (int)((1u << 16) |                                // data_size = 2B
                (pad ? (1u << 20) : 0u) |                   // pad_enable
                (padi << 22) | (pada << 25));               // interval / amount
  g1[1] = (int)((td0 & 0xFFFFu) << 16);                     // tensor_dim0 lo
  g1[2] = (int)((td0 >> 16) | ((td1 & 0xFFFFu) << 16));     // dim0 hi | dim1 lo
  g1[3] = (int)((td1 >> 16) | (tile0 << 16));               // dim1 hi | tile0
  g1[4] = (int)(tile1 & 0xFFFFu);                           // tile1 | tile2=0
  g1[5] = (int)(unsigned)(stride0 & 0xFFFFFFFFull);         // dim0_stride lo
  g1[6] = (int)(unsigned)((stride0 >> 32) & 0xFFFFull);     // dim0_stride hi
  g1[7] = 0;                                                // dim1_stride (2D)
  const i32x4 z = {0, 0, 0, 0};
#if defined(__clang_major__) && (__clang_major__ >= 23)
  const i32x8 z8 = {0, 0, 0, 0, 0, 0, 0, 0};
  __builtin_amdgcn_tensor_load_to_lds(g0, g1, z, z, z8, 0);
#else
  __builtin_amdgcn_tensor_load_to_lds(g0, g1, z, z, 0);
#endif
}

__device__ __forceinline__ unsigned lds_off(const void* p) {
  return (unsigned)(unsigned long long)p;  // generic LDS ptr: low 32b = offset
}

// ---------------------------------------------------------------------------
// Kernel 1: fused 1x1-conv (+bias, folded-BN affine, ReLU) for Q, K, V.
//   out(o,n) = relu(g[o]*(W[o,:]·in[:,n] + b[o]) + beta[o]),  o = h*D + d
// bf16 scratch layouts chosen so attention fragments are pure b128 loads:
//   Qb: (bh, n, d) row-major   (A-fragments, d-contiguous)
//   Kb: (bh, n, d) row-major   (B-fragments of Q·K^T: col=n, k=d contiguous)
//   Vt: (bh, d, n) transposed  (B-fragments of P·V:   col=d, k=n contiguous)
// Grid: (OC/128, N/64, 3*B). Block: 256 threads (8 waves stacked along o).
// Staging uses unrolled float4 (b128) loads; ALL results are repacked through
// LDS into coalesced b128 stores (each WG's o-range stays within one head).
// ---------------------------------------------------------------------------
__global__ __launch_bounds__(256) void proj_kernel(
    const float* __restrict__ x, const float* __restrict__ y,
    const float* __restrict__ Wq, const float* __restrict__ bq,
    const float* __restrict__ gq, const float* __restrict__ betaq,
    const float* __restrict__ Wk, const float* __restrict__ bk,
    const float* __restrict__ gk, const float* __restrict__ betak,
    const float* __restrict__ Wv, const float* __restrict__ bv,
    const float* __restrict__ gv, const float* __restrict__ betav,
    unsigned short* __restrict__ Qb, unsigned short* __restrict__ Kb,
    unsigned short* __restrict__ Vt) {
  const int mb  = blockIdx.z;           // mat*B + b
  const int mat = mb / BB;
  const int b   = mb % BB;

  const float *W, *bias, *g, *beta, *in;
  if (mat == 0)      { W = Wq; bias = bq; g = gq; beta = betaq; in = x; }
  else if (mat == 1) { W = Wk; bias = bk; g = gk; beta = betak; in = y; }
  else               { W = Wv; bias = bv; g = gv; beta = betav; in = y; }

  const int lane = threadIdx.x & 31;
  const int wave = threadIdx.x >> 5;
  const int lo   = lane & 15;
  const int hi   = lane >> 4;
  const int o0   = blockIdx.x * 128 + wave * 16;  // this wave's 16 output rows
  const int n0   = blockIdx.y * 64;               // 64 output columns per WG

  // One LDS arena, phase-aliased:
  //  staging    : xs = [64 n][40] (2560 halves), wls = [128 o][40] (5120)
  //  repack Q/K : [64 n][136 pad]  (8704 halves, 16B-aligned rows)
  //  repack V   : [128 d][72 pad]  (9216 halves, 16B-aligned rows)
  __shared__ unsigned short smem[9216];
  unsigned short* xs  = smem;
  unsigned short* wls = smem + 64 * 40;

  v8f acc[4] = {};

  for (int k0 = 0; k0 < CC; k0 += 32) {
    __syncthreads();
    // x tile: 32(k) x 64(n); float4 along n, coalesced; 2 per thread.
#pragma unroll
    for (int it = 0; it < 2; ++it) {
      const int idx = threadIdx.x + it * 256;     // 0..511
      const int kk  = idx >> 4;                   // 0..31
      const int nn  = (idx & 15) * 4;             // 0..60
      const f32x4 f = *(const f32x4*)&in[(b * CC + k0 + kk) * NN + n0 + nn];
      xs[(nn + 0) * 40 + kk] = f2bf(f[0]);
      xs[(nn + 1) * 40 + kk] = f2bf(f[1]);
      xs[(nn + 2) * 40 + kk] = f2bf(f[2]);
      xs[(nn + 3) * 40 + kk] = f2bf(f[3]);
    }
    // W tile: 128(o) x 32(k); float4 along k, coalesced; 4 per thread.
#pragma unroll
    for (int it = 0; it < 4; ++it) {
      const int idx = threadIdx.x + it * 256;     // 0..1023
      const int oo  = idx >> 3;                   // 0..127
      const int kk  = (idx & 7) * 4;              // 0..28
      const f32x4 f = *(const f32x4*)&W[(blockIdx.x * 128 + oo) * CC + k0 + kk];
      u32x2 p;
      p[0] = pack2bf(f[0], f[1]);
      p[1] = pack2bf(f[2], f[3]);
      *(u32x2*)&wls[oo * 40 + kk] = p;            // ds_store_b64
    }
    __syncthreads();

    Frag a;
    loadA(a, &wls[(wave * 16 + lo) * 40], hi);    // 2x ds_load_b128

#pragma unroll
    for (int nb = 0; nb < 4; ++nb) {
      Frag xb;
      loadB(xb, &xs[(nb * 16 + lo) * 40], hi);    // 2x ds_load_b128
      acc[nb] = __builtin_amdgcn_wmma_f32_16x16x32_bf16(
          false, a.v, false, xb.v, (short)0, acc[nb], false, false);
    }
  }

  // ---- Epilogue: bias + affine + ReLU, repack via LDS ----
  // Hoist per-row affine params: only 8 distinct o per lane (independent of nb).
  float biasr[8], gr[8], betar[8];
#pragma unroll
  for (int r = 0; r < 8; ++r) {
    const int o = o0 + r + 8 * hi;
    biasr[r] = bias[o];
    gr[r]    = g[o];
    betar[r] = beta[o];
  }

  __syncthreads();                                // staging reads all done
#pragma unroll
  for (int nb = 0; nb < 4; ++nb) {
#pragma unroll
    for (int r = 0; r < 8; ++r) {
      const int oloc = wave * 16 + r + 8 * hi;    // 0..127 (o - oBase)
      const int nloc = nb * 16 + lo;              // 0..63  (n - n0)
      float z = acc[nb][r] + biasr[r];
      z = fmaxf(gr[r] * z + betar[r], 0.0f);
      const unsigned short h = f2bf(z);
      // mat 0/1: [n][o] rows of 136; mat 2: [d][n] rows of 72 (uniform select)
      const int pidx = (mat == 2) ? (oloc * 72 + nloc) : (nloc * 136 + oloc);
      smem[pidx] = h;
    }
  }
  __syncthreads();

  const int oBase = blockIdx.x * 128;             // single head per WG
  const size_t bh = (size_t)b * HH + (oBase >> 8);
  const int d0    = oBase & 255;
  if (mat == 2) {
    // 128 (d) rows x 64 halves -> coalesced b128 stores, 4 per thread.
#pragma unroll
    for (int it = 0; it < 4; ++it) {
      const int idx  = threadIdx.x + it * 256;    // 0..1023
      const int rowd = idx >> 3;                  // 0..127
      const int c8   = (idx & 7) * 8;             // halves 0..56
      const u32x4 vdat = *(const u32x4*)&smem[rowd * 72 + c8];
      *(u32x4*)&Vt[(bh * DD + d0 + rowd) * NN + n0 + c8] = vdat;
    }
  } else {
    unsigned short* dst = (mat == 0) ? Qb : Kb;
    // 64 (n) rows x 128 halves -> coalesced b128 stores, 4 per thread.
#pragma unroll
    for (int it = 0; it < 4; ++it) {
      const int idx  = threadIdx.x + it * 256;    // 0..1023
      const int rown = idx >> 4;                  // 0..63
      const int c8   = (idx & 15) * 8;            // halves 0..120
      const u32x4 vdat = *(const u32x4*)&smem[rown * 136 + c8];
      *(u32x4*)&dst[(bh * NN + n0 + rown) * DD + d0 + c8] = vdat;
    }
  }
}

// ---------------------------------------------------------------------------
// Kernel 2: flash attention. Grid: (N/128, B*H), 256 threads = 8 waves.
// Each wave: 16 query rows; Q as 8 bf16 A-fragments and O as 16 f32
// C-fragments in registers. K/V tiles (32 keys) are DMA'd into LDS by the
// Tensor Data Mover (double-buffered, producer = wave 0, TENSORcnt +
// workgroup barrier), with HW row padding so ds_load_b128 fragment reads are
// bank-conflict-free. TDM(next tile) overlaps the 32 WMMAs of this tile.
// ---------------------------------------------------------------------------
__global__ __launch_bounds__(256) void attn_kernel(
    const unsigned short* __restrict__ Qb, const unsigned short* __restrict__ Kb,
    const unsigned short* __restrict__ Vt, float* __restrict__ out) {
  const size_t bh  = blockIdx.y;
  const int   wave = threadIdx.x >> 5;
  const int   lane = threadIdx.x & 31;
  const int   lo   = lane & 15;
  const int   hi   = lane >> 4;
  const int   i0   = blockIdx.x * 128 + wave * 16;

  // K tile: 32 rows x 256 halves, TDM-padded 512B->528B rows (132 dwords).
  // V tile: 256 rows x  32 halves, TDM-padded  64B-> 80B rows ( 20 dwords).
  __shared__ unsigned short kbuf[2][32 * 264];
  __shared__ unsigned short vbuf[2][256 * 40];
  __shared__ unsigned short pstage[8][16 * 40];  // per-wave P transpose

  const unsigned short* Kbh = Kb + bh * NN * DD;
  const unsigned short* Vbh = Vt + bh * DD * NN;

  // Preload Q block: 8 A-fragments, 2x global b128 each.
  Frag qa[8];
  const unsigned short* Qrow = Qb + (bh * NN + i0 + lo) * DD;
#pragma unroll
  for (int t = 0; t < 8; ++t) loadA(qa[t], Qrow + 32 * t, hi);

  v8f o_acc[16] = {};
  float m[8], l[8];
#pragma unroll
  for (int r = 0; r < 8; ++r) { m[r] = -__builtin_inff(); l[r] = 0.0f; }

  // Issue tile 0 (K: 256x32 tile of (256 x 2048)-strided tensor; V: 32x256).
  if (wave == 0) {
    tdm_load_2d(lds_off(kbuf[0]), Kbh, DD, NN, DD, 32, DD, 6u, 3u, true);
    tdm_load_2d(lds_off(vbuf[0]), Vbh, NN, DD, 32, DD, NN, 3u, 3u, true);
  }

  for (int j0 = 0; j0 < NN; j0 += 32) {
    const int p = (j0 >> 5) & 1;
    if (wave == 0) __builtin_amdgcn_s_wait_tensorcnt(0);  // tile ready (wave0)
    __syncthreads();                                      // broadcast + reuse gate
    if (wave == 0 && j0 + 32 < NN) {                      // prefetch next tile
      tdm_load_2d(lds_off(kbuf[1 - p]), Kbh + (size_t)(j0 + 32) * DD,
                  DD, NN, DD, 32, DD, 6u, 3u, true);
      tdm_load_2d(lds_off(vbuf[1 - p]), Vbh + (j0 + 32),
                  NN, DD, 32, DD, NN, 3u, 3u, true);
    }
    const unsigned short* kt = kbuf[p];
    const unsigned short* vt = vbuf[p];

    // ---- S = Q · K^T  (two 16x16 C-fragments covering 32 keys) ----
    v8f s0 = {}, s1 = {};
#pragma unroll
    for (int t = 0; t < 8; ++t) {
      Frag kb0, kb1;
      loadB(kb0, kt + lo * 264 + 32 * t, hi);        // 2x ds_load_b128
      loadB(kb1, kt + (16 + lo) * 264 + 32 * t, hi);
      s0 = __builtin_amdgcn_wmma_f32_16x16x32_bf16(false, qa[t].v, false, kb0.v,
                                                   (short)0, s0, false, false);
      s1 = __builtin_amdgcn_wmma_f32_16x16x32_bf16(false, qa[t].v, false, kb1.v,
                                                   (short)0, s1, false, false);
    }

    // ---- online softmax over this 32-key tile ----
    float scale[8];
#pragma unroll
    for (int r = 0; r < 8; ++r) {
      float tm = fmaxf(s0[r], s1[r]);
#pragma unroll
      for (int off = 8; off >= 1; off >>= 1)
        tm = fmaxf(tm, __shfl_xor(tm, off, 32));     // 16-lane row group
      const float mn = fmaxf(m[r], tm);
      const float sc = __expf(m[r] - mn);
      const float p0 = __expf(s0[r] - mn);
      const float p1 = __expf(s1[r] - mn);
      s0[r] = p0; s1[r] = p1;
      float ts = p0 + p1;
#pragma unroll
      for (int off = 8; off >= 1; off >>= 1)
        ts += __shfl_xor(ts, off, 32);
      l[r] = l[r] * sc + ts;
      m[r] = mn;
      scale[r] = sc;
    }
#pragma unroll
    for (int dt = 0; dt < 16; ++dt)
#pragma unroll
      for (int r = 0; r < 8; ++r) o_acc[dt][r] *= scale[r];

    // ---- P (C-layout) -> bf16 A-fragment via per-wave LDS transpose ----
    unsigned short* ps = pstage[wave];
#pragma unroll
    for (int r = 0; r < 8; ++r) {
      const int row = r + 8 * hi;
      ps[row * 40 + lo]      = f2bf(s0[r]);
      ps[row * 40 + 16 + lo] = f2bf(s1[r]);
    }
    asm volatile("s_wait_dscnt 0" ::: "memory");     // own-wave LDS RAW
    Frag pa;
    loadA(pa, ps + lo * 40, hi);                     // 2x ds_load_b128

    // ---- O += P · V  (16 d-tiles) ----
#pragma unroll
    for (int dt = 0; dt < 16; ++dt) {
      Frag vb;
      loadB(vb, vt + (dt * 16 + lo) * 40, hi);       // 2x ds_load_b128
      o_acc[dt] = __builtin_amdgcn_wmma_f32_16x16x32_bf16(
          false, pa.v, false, vb.v, (short)0, o_acc[dt], false, false);
    }
    __syncthreads();                                 // done reading buf[p]
  }

  // ---- normalize and write (B,H,N,1,D) f32, coalesced along d ----
#pragma unroll
  for (int dt = 0; dt < 16; ++dt) {
#pragma unroll
    for (int r = 0; r < 8; ++r) {
      const int row = i0 + r + 8 * hi;
      const int d   = dt * 16 + lo;
      out[(bh * NN + row) * DD + d] = o_acc[dt][r] / l[r];
    }
  }
}

// ---------------------------------------------------------------------------
extern "C" void kernel_launch(void* const* d_in, const int* in_sizes, int n_in,
                              void* d_out, int out_size, void* d_ws, size_t ws_size,
                              hipStream_t stream) {
  const float* x     = (const float*)d_in[0];
  const float* y     = (const float*)d_in[1];
  const float* Wq    = (const float*)d_in[2];
  const float* bq    = (const float*)d_in[3];
  const float* gq    = (const float*)d_in[4];
  const float* betaq = (const float*)d_in[5];
  const float* Wk    = (const float*)d_in[6];
  const float* bk    = (const float*)d_in[7];
  const float* gk    = (const float*)d_in[8];
  const float* betak = (const float*)d_in[9];
  const float* Wv    = (const float*)d_in[10];
  const float* bv    = (const float*)d_in[11];
  const float* gv    = (const float*)d_in[12];
  const float* betav = (const float*)d_in[13];

  const size_t elems = (size_t)BB * HH * NN * DD;      // 16.8M bf16 each
  unsigned short* Qb = (unsigned short*)d_ws;
  unsigned short* Kb = Qb + elems;
  unsigned short* Vt = Kb + elems;

  dim3 pg(OCC / 128, NN / 64, 3 * BB);
  proj_kernel<<<pg, 256, 0, stream>>>(x, y, Wq, bq, gq, betaq, Wk, bk, gk,
                                      betak, Wv, bv, gv, betav, Qb, Kb, Vt);

  dim3 ag(NN / 128, BB * HH);
  attn_kernel<<<ag, 256, 0, stream>>>(Qb, Kb, Vt, (float*)d_out);
}
